// QuantumLSTM_86174223827708
// MI455X (gfx1250) — matrix operand order repaired and checked by reading, not compiled, not run on hardware
//
#include <hip/hip_runtime.h>
#include <hip/hip_bf16.h>

#ifndef __has_builtin
#define __has_builtin(x) 0
#endif

typedef __attribute__((ext_vector_type(16))) __bf16 v16bf;
typedef __attribute__((ext_vector_type(8)))  __bf16 v8bf;
typedef __attribute__((ext_vector_type(8)))  float  v8f;
typedef int v4i_ __attribute__((vector_size(16)));
typedef __attribute__((address_space(1))) v4i_* g4i_t;   // global v4i*
typedef __attribute__((address_space(3))) v4i_* l4i_t;   // lds v4i*

#define BB 32
#define SS 2048
#define IIN 128
#define HH 256
#define QQ 16
#define MM 64
#define TT (BB * SS)     // 65536 tokens
#define G4 (4 * HH)      // 1024 gates

#if defined(__HIP_DEVICE_COMPILE__)
#if __has_builtin(__builtin_amdgcn_global_load_async_to_lds_b128)
#define HAVE_ASYNC 1
#else
#define HAVE_ASYNC 0
#endif
#if __has_builtin(__builtin_amdgcn_s_wait_asynccnt)
#define WAIT_ASYNC() __builtin_amdgcn_s_wait_asynccnt(0)
#else
#define WAIT_ASYNC() asm volatile("s_wait_asynccnt 0" ::: "memory")
#endif
#endif

__device__ inline v16bf make16(v8bf lo, v8bf hi) {
    v16bf r;
#pragma unroll
    for (int i = 0; i < 8; ++i) { r[i] = lo[i]; r[8 + i] = hi[i]; }
    return r;
}
__device__ inline v8bf zero8bf() {
    v8bf z;
#pragma unroll
    for (int i = 0; i < 8; ++i) z[i] = (__bf16)0.f;
    return z;
}
__device__ inline float sigmoidf_(float x) { return 1.f / (1.f + __expf(-x)); }

// ---------------------------------------------------------------- utilities
__global__ void k_f2bf(const float* __restrict__ src, __bf16* __restrict__ dst, int n) {
    for (int i = blockIdx.x * blockDim.x + threadIdx.x; i < n; i += gridDim.x * blockDim.x)
        dst[i] = (__bf16)src[i];
}

__global__ void k_zero32(unsigned* __restrict__ p, int nwords) {
    for (int i = blockIdx.x * blockDim.x + threadIdx.x; i < nwords; i += gridDim.x * blockDim.x)
        p[i] = 0u;
}

__global__ void k_normkeys(const float* __restrict__ keys, float* __restrict__ kn) {
    int m = threadIdx.x;  // 64 threads
    if (m >= MM) return;
    float s = 0.f;
#pragma unroll
    for (int i = 0; i < QQ; ++i) { float v = keys[m * QQ + i]; s += v * v; }
    float inv = 1.f / (sqrtf(s) + 1e-8f);
#pragma unroll
    for (int i = 0; i < QQ; ++i) kn[m * QQ + i] = keys[m * QQ + i] * inv;
}

// ---------------------------------------------------------------- WMMA GEMM
// C[M,N] = A[M,K](bf16,row-major) * W[N,K]^T (bf16,row-major) + b1[N] + b2[N]
// Block: 256 threads = 8 waves. Each wave owns MT 16-row M-tiles and sweeps N
// in chunks of NT column-tiles; fully unrolled, branch-free WMMA inner loop.
// Per k-step: MT A-frag loads + NT B-frag loads feed MT*NT WMMAs.
template <int MT, int NT>
__global__ void k_gemm_bf16(const __bf16* __restrict__ A, const __bf16* __restrict__ W,
                            const float* __restrict__ b1, const float* __restrict__ b2,
                            float* __restrict__ C, int M, int N, int K) {
    const int lane = threadIdx.x & 31;
    const int wave = threadIdx.x >> 5;
    const int mt0 = (blockIdx.x * 8 + wave) * MT;
    if (mt0 * 16 >= M) return;
    const int Nt = N >> 4, Kt = K >> 5;

    const int aoff = (lane & 16) ? 8 : 0;       // A: lanes 16-31 start at K+8
    const int boff = (lane & 16) ? 8 : 0;       // B: lanes 16-31 start at K+8
    const int col15 = lane & 15;
    const int rowoff = (lane & 16) ? 8 : 0;

    const __bf16* arowp[MT];
#pragma unroll
    for (int mi = 0; mi < MT; ++mi)
        arowp[mi] = A + (size_t)((mt0 + mi) * 16 + col15) * K + aoff;

    for (int nt0 = 0; nt0 < Nt; nt0 += NT) {
        v8f acc[MT][NT];
#pragma unroll
        for (int mi = 0; mi < MT; ++mi)
#pragma unroll
            for (int j = 0; j < NT; ++j) acc[mi][j] = {};

        for (int ks = 0; ks < Kt; ++ks) {
            v16bf a[MT];
#pragma unroll
            for (int mi = 0; mi < MT; ++mi) {
                const __bf16* ap = arowp[mi] + ks * 32;
                a[mi] = make16(*(const v8bf*)ap, *(const v8bf*)(ap + 16));
            }
#pragma unroll
            for (int j = 0; j < NT; ++j) {
                const __bf16* bp = W + (size_t)((nt0 + j) * 16 + col15) * K + ks * 32 + boff;
            v16bf b = make16(*(const v8bf*)bp, *(const v8bf*)(bp + 16));
#pragma unroll
                for (int mi = 0; mi < MT; ++mi)
                    acc[mi][j] = __builtin_amdgcn_wmma_f32_16x16x32_bf16(
                        false, a[mi], false, b, (short)0, acc[mi][j], false, false);
            }
        }
#pragma unroll
        for (int j = 0; j < NT; ++j) {
            int col = (nt0 + j) * 16 + col15;
            float bias = (b1 ? b1[col] : 0.f) + (b2 ? b2[col] : 0.f);
#pragma unroll
            for (int mi = 0; mi < MT; ++mi) {
                int mbase = (mt0 + mi) * 16 + rowoff;
#pragma unroll
                for (int r = 0; r < 8; ++r)
                    C[(size_t)(mbase + r) * N + col] = acc[mi][j][r] + bias;
            }
        }
    }
}

// ---------------------------------------------------------------- LSTM recurrence
// 16 persistent workgroups; WG g owns hidden columns [16g,16g+16).
// LDS: its 64x256 bf16 W_hh slice, the full 32x256 bf16 h, fp32 c slice.
// Per step: 16 WMMA ops/wave -> gates, elementwise LSTM cell, h slice to L2,
// grid arrive/spin barrier, async reload of full h into LDS.
#define NWG 16
#define RT 128
#define HSTR 264   // padded bf16 row stride (132 dwords -> conflict-free b128)
#define GSTR 68

__global__ void k_lstm_recur(const float* __restrict__ xg, const __bf16* __restrict__ Whh,
                             __bf16* __restrict__ hex0, __bf16* __restrict__ hex1,
                             __bf16* __restrict__ hout, unsigned* __restrict__ ctr) {
    __shared__ __bf16 Whh_s[64 * HSTR];
    __shared__ __bf16 h_s[32 * HSTR];
    __shared__ float  gates_s[32 * GSTR];
    __shared__ float  c_s[32 * 17];

    const int tid = threadIdx.x;
    const int g = blockIdx.x;
    const int wave = tid >> 5, lane = tid & 31;

    // Stage W_hh slice: local row lr -> global gate row (lr/16)*256 + 16g + lr%16
    for (int i = tid; i < 64 * 32; i += RT) {
        int lr = i >> 5, ch = i & 31;
        int grow = ((lr >> 4) << 8) + g * 16 + (lr & 15);
        *(v8bf*)&Whh_s[lr * HSTR + ch * 8] = *(const v8bf*)(Whh + (size_t)grow * HH + ch * 8);
    }
    for (int i = tid; i < 32 * 32; i += RT) {
        int r = i >> 5, ch = i & 31;
        *(v8bf*)&h_s[r * HSTR + ch * 8] = zero8bf();
    }
    for (int i = tid; i < 32 * 17; i += RT) c_s[i] = 0.f;
    __syncthreads();

    const int mt = wave & 1;
    const int ntA = (wave >> 1) * 2, ntB = ntA + 1;
    const int arow = mt * 16 + (lane & 15);
    const int aoff = (lane & 16) ? 8 : 0;
    const int boff = (lane & 16) ? 8 : 0;
    const int brA = ntA * 16 + (lane & 15);
    const int brB = ntB * 16 + (lane & 15);
    const int col15 = lane & 15;
    const int rowoff = (lane & 16) ? 8 : 0;

    for (int t = 0; t < SS; ++t) {
        v8f accA = {}, accB = {};
#pragma unroll
        for (int ks = 0; ks < 8; ++ks) {
            int ak = ks * 32 + aoff;
            v16bf a = make16(*(const v8bf*)&h_s[arow * HSTR + ak],
                             *(const v8bf*)&h_s[arow * HSTR + ak + 16]);
            int bk = ks * 32 + boff;
            v16bf b0 = make16(*(const v8bf*)&Whh_s[brA * HSTR + bk],
                              *(const v8bf*)&Whh_s[brA * HSTR + bk + 16]);
            accA = __builtin_amdgcn_wmma_f32_16x16x32_bf16(false, a, false, b0, (short)0, accA, false, false);
            v16bf b1 = make16(*(const v8bf*)&Whh_s[brB * HSTR + bk],
                              *(const v8bf*)&Whh_s[brB * HSTR + bk + 16]);
            accB = __builtin_amdgcn_wmma_f32_16x16x32_bf16(false, a, false, b1, (short)0, accB, false, false);
        }
#pragma unroll
        for (int r = 0; r < 8; ++r) {
            gates_s[(mt * 16 + rowoff + r) * GSTR + ntA * 16 + col15] = accA[r];
            gates_s[(mt * 16 + rowoff + r) * GSTR + ntB * 16 + col15] = accB[r];
        }
        __syncthreads();

        __bf16* hexw = (t & 1) ? hex1 : hex0;
#pragma unroll
        for (int j = 0; j < 4; ++j) {
            int idx = tid + j * RT;          // 512 cell updates: 32 batch x 16 cols
            int b = idx & 31, hc = idx >> 5;
            int gcol = g * 16 + hc;
            size_t xrow = (size_t)(b * SS + t) * G4;
            float iv = sigmoidf_(gates_s[b * GSTR + 0 * 16 + hc] + xg[xrow + 0 * HH + gcol]);
            float fv = sigmoidf_(gates_s[b * GSTR + 1 * 16 + hc] + xg[xrow + 1 * HH + gcol]);
            float gv = tanhf    (gates_s[b * GSTR + 2 * 16 + hc] + xg[xrow + 2 * HH + gcol]);
            float ov = sigmoidf_(gates_s[b * GSTR + 3 * 16 + hc] + xg[xrow + 3 * HH + gcol]);
            float c = fv * c_s[b * 17 + hc] + iv * gv;
            c_s[b * 17 + hc] = c;
            __bf16 hb = (__bf16)(ov * tanhf(c));
            hexw[b * HH + gcol] = hb;
            hout[(size_t)(b * SS + t) * HH + gcol] = hb;
        }
        __syncthreads();
        __threadfence();
        if (tid == 0) {
            atomicAdd(ctr, 1u);
            unsigned target = (unsigned)(t + 1) * NWG;
            while (__hip_atomic_load(ctr, __ATOMIC_RELAXED, __HIP_MEMORY_SCOPE_AGENT) < target)
                __builtin_amdgcn_s_sleep(2);
        }
        __syncthreads();
        __threadfence();

        const __bf16* hexr = (t & 1) ? hex1 : hex0;
#if HAVE_ASYNC
        for (int i = tid; i < 32 * 32; i += RT) {
            int r = i >> 5, ch = i & 31;
            __builtin_amdgcn_global_load_async_to_lds_b128(
                (g4i_t)(hexr + r * HH + ch * 8),
                (l4i_t)&h_s[r * HSTR + ch * 8], 0, 0);
        }
        WAIT_ASYNC();
#else
        for (int i = tid; i < 32 * 32; i += RT) {
            int r = i >> 5, ch = i & 31;
            *(v8bf*)&h_s[r * HSTR + ch * 8] = *(const v8bf*)(hexr + r * HH + ch * 8);
        }
#endif
        __syncthreads();
    }
}

// ---------------------------------------------------------------- quantum knn read
// one thread per token: tanh(q), normalize, cosine sims vs 64 normalized keys,
// top-3, weighted read of mem_vals, emit combined bf16 row [0.7*h | 0.3*enh].
__global__ void k_quantum(const float* __restrict__ qbuf, const __bf16* __restrict__ lstm_bf,
                          const float* __restrict__ kn, const float* __restrict__ vals,
                          const float* __restrict__ cwp, const float* __restrict__ qwp,
                          __bf16* __restrict__ comb) {
    __shared__ float kn_s[MM * QQ];
    int tid = threadIdx.x;
    for (int i = tid; i < MM * QQ; i += 256) kn_s[i] = kn[i];
    __syncthreads();
    int token = blockIdx.x * 256 + tid;
    if (token >= TT) return;

    float q[QQ]; float ss = 0.f;
#pragma unroll
    for (int i = 0; i < QQ; ++i) {
        float v = tanhf(qbuf[(size_t)token * QQ + i]);
        q[i] = v; ss += v * v;
    }
    float inv = 1.f / (sqrtf(ss) + 1e-8f);

    float v0 = -1e30f, v1 = -1e30f, v2 = -1e30f; int i0 = 0, i1 = 0, i2 = 0;
    for (int m = 0; m < MM; ++m) {
        float s = 0.f;
#pragma unroll
        for (int i = 0; i < QQ; ++i) s += q[i] * kn_s[m * QQ + i];
        s *= inv;
        if (s > v0)      { v2 = v1; i2 = i1; v1 = v0; i1 = i0; v0 = s; i0 = m; }
        else if (s > v1) { v2 = v1; i2 = i1; v1 = s; i1 = m; }
        else if (s > v2) { v2 = s; i2 = m; }
    }
    float tot = v0 + v1 + v2;
    float cw = *cwp, qw = *qwp;
    float sc = (tot > 0.f) ? (qw / ((tot == 0.f) ? 1.f : tot)) : 0.f;
    const float* r0 = vals + i0 * HH;
    const float* r1 = vals + i1 * HH;
    const float* r2 = vals + i2 * HH;
    size_t cbase = (size_t)token * (2 * HH);
    size_t lbase = (size_t)token * HH;
    for (int h = 0; h < HH; ++h) {
        comb[cbase + h]      = (__bf16)(cw * (float)lstm_bf[lbase + h]);
        comb[cbase + HH + h] = (__bf16)(sc * (v0 * r0[h] + v1 * r1[h] + v2 * r2[h]));
    }
}

// ---------------------------------------------------------------- host side
static inline size_t alup(size_t x) { return (x + 255) & ~(size_t)255; }

extern "C" void kernel_launch(void* const* d_in, const int* in_sizes, int n_in,
                              void* d_out, int out_size, void* d_ws, size_t ws_size,
                              hipStream_t stream) {
    (void)in_sizes; (void)n_in; (void)out_size; (void)ws_size;
    const float* x     = (const float*)d_in[0];
    const float* Wih0  = (const float*)d_in[1];
    const float* Whh0  = (const float*)d_in[2];
    const float* bih0  = (const float*)d_in[3];
    const float* bhh0  = (const float*)d_in[4];
    const float* Wih1  = (const float*)d_in[5];
    const float* Whh1  = (const float*)d_in[6];
    const float* bih1  = (const float*)d_in[7];
    const float* bhh1  = (const float*)d_in[8];
    const float* Wcq   = (const float*)d_in[9];
    const float* bcq   = (const float*)d_in[10];
    const float* mkeys = (const float*)d_in[11];
    const float* mvals = (const float*)d_in[12];
    const float* Wout  = (const float*)d_in[13];
    const float* bout  = (const float*)d_in[14];
    const float* cw    = (const float*)d_in[15];
    const float* qw    = (const float*)d_in[16];
    float* out = (float*)d_out;

    char* w = (char*)d_ws; size_t cur = 0;
    auto take = [&](size_t bytes) { char* p = w + cur; cur = alup(cur + bytes); return p; };
    __bf16* xbf     = (__bf16*)take((size_t)TT * IIN * 2);
    __bf16* wih0b   = (__bf16*)take((size_t)G4 * IIN * 2);
    __bf16* whh0b   = (__bf16*)take((size_t)G4 * HH * 2);
    __bf16* wih1b   = (__bf16*)take((size_t)G4 * HH * 2);
    __bf16* whh1b   = (__bf16*)take((size_t)G4 * HH * 2);
    __bf16* wcqb    = (__bf16*)take((size_t)QQ * HH * 2);
    __bf16* woutb   = (__bf16*)take((size_t)HH * 2 * HH * 2);
    float*  xg      = (float*)take((size_t)TT * G4 * 4);
    __bf16* h1bf    = (__bf16*)take((size_t)TT * HH * 2);
    __bf16* lstmbf  = (__bf16*)take((size_t)TT * HH * 2);
    float*  qbuf    = (float*)take((size_t)TT * QQ * 4);
    float*  kn      = (float*)take((size_t)MM * QQ * 4);
    __bf16* combbf  = (__bf16*)take((size_t)TT * 2 * HH * 2);
    __bf16* hex0    = (__bf16*)take((size_t)BB * HH * 2);
    __bf16* hex1    = (__bf16*)take((size_t)BB * HH * 2);
    unsigned* ctr   = (unsigned*)take(256);

    // bf16 conversions
    k_f2bf<<<512, 256, 0, stream>>>(x, xbf, TT * IIN);
    k_f2bf<<<64, 256, 0, stream>>>(Wih0, wih0b, G4 * IIN);
    k_f2bf<<<64, 256, 0, stream>>>(Whh0, whh0b, G4 * HH);
    k_f2bf<<<64, 256, 0, stream>>>(Wih1, wih1b, G4 * HH);
    k_f2bf<<<64, 256, 0, stream>>>(Whh1, whh1b, G4 * HH);
    k_f2bf<<<16, 256, 0, stream>>>(Wcq, wcqb, QQ * HH);
    k_f2bf<<<64, 256, 0, stream>>>(Wout, woutb, HH * 2 * HH);
    k_normkeys<<<1, 64, 0, stream>>>(mkeys, kn);

    // layer 0: xg0 = x*Wih0^T + b, then recurrence
    k_gemm_bf16<2, 4><<<TT / 256, 256, 0, stream>>>(xbf, wih0b, bih0, bhh0, xg, TT, G4, IIN);
    k_zero32<<<33, 256, 0, stream>>>((unsigned*)hex0, (2 * BB * HH * 2 + 256) / 4);
    k_lstm_recur<<<NWG, RT, 0, stream>>>(xg, whh0b, hex0, hex1, h1bf, ctr);

    // layer 1: xg1 = h1*Wih1^T + b, then recurrence
    k_gemm_bf16<2, 4><<<TT / 256, 256, 0, stream>>>(h1bf, wih1b, bih1, bhh1, xg, TT, G4, HH);
    k_zero32<<<33, 256, 0, stream>>>((unsigned*)hex0, (2 * BB * HH * 2 + 256) / 4);
    k_lstm_recur<<<NWG, RT, 0, stream>>>(xg, whh1b, hex0, hex1, lstmbf, ctr);

    // q = lstm*Wcq^T + bcq ; quantum knn read -> combined(bf16)
    k_gemm_bf16<2, 1><<<TT / 256, 256, 0, stream>>>(lstmbf, wcqb, bcq, nullptr, qbuf, TT, QQ, HH);
    k_quantum<<<TT / 256, 256, 0, stream>>>(qbuf, lstmbf, kn, mvals, cw, qw, combbf);

    // out = combined*Wout^T + bout
    k_gemm_bf16<2, 4><<<TT / 256, 256, 0, stream>>>(combbf, woutb, bout, nullptr, out, TT, HH, 2 * HH);
}
